// EGATencoderlayer_41601053229524
// MI455X (gfx1250) — matrix operand order repaired and checked by reading, not compiled, not run on hardware
//
#include <hip/hip_runtime.h>
#include <hip/hip_bf16.h>
#include <math.h>

#define Bn 8
#define Nn 512
#define Dn 768
#define Hn 8
#define En 96
#define LEAKY 0.2f
#define NEGV -99999.0f

typedef __bf16 bf16;
typedef __attribute__((ext_vector_type(16))) __bf16 v16bf;
typedef __attribute__((ext_vector_type(8)))  __bf16 v8bf;
typedef __attribute__((ext_vector_type(8)))  float  v8f;

// ---------------------------------------------------------------------------
// WMMA fragment loaders (bf16, 16x16x32, wave32 layouts per CDNA5 ISA 7.12.2)
// A: row-major [M x K], lda = K. lane<16: row m0+lane, K {0..7, 16..23};
//    lane>=16: row m0+lane-16, K {8..15, 24..31}.
// B: supplied TRANSPOSED row-major [Ncols x K], ldb = K. lane<16: col n0+lane,
//    K {0..7, 16..23}; lane>=16: col n0+lane-16, K {8..15, 24..31}.
// ---------------------------------------------------------------------------
__device__ __forceinline__ v16bf load_a_frag(const bf16* A, int lda, int m0, int k0, int lane) {
  const int half = lane >> 4;
  const int m    = m0 + (lane & 15);
  const bf16* p  = A + (size_t)m * lda + k0 + half * 8;
  v8bf lo = *(const v8bf*)p;        // K offs 0..7  (or 8..15)
  v8bf hi = *(const v8bf*)(p + 16); // K offs 16..23 (or 24..31)
  v16bf r;
#pragma unroll
  for (int i = 0; i < 8; ++i) { r[i] = lo[i]; r[i + 8] = hi[i]; }
  return r;
}

__device__ __forceinline__ v16bf load_b_frag(const bf16* BT, int ldb, int n0, int k0, int lane) {
  const int half = lane >> 4;
  const int n    = n0 + (lane & 15);
  const bf16* p  = BT + (size_t)n * ldb + k0 + half * 8;
  v8bf lo = *(const v8bf*)p;
  v8bf hi = *(const v8bf*)(p + 16);
  v16bf r;
#pragma unroll
  for (int i = 0; i < 8; ++i) { r[i] = lo[i]; r[i + 8] = hi[i]; }
  return r;
}

__device__ __forceinline__ v8f wmma_bf16(v16bf a, v16bf b, v8f c) {
  return __builtin_amdgcn_wmma_f32_16x16x32_bf16(false, a, false, b, (short)0, c, false, false);
}

// ---------------------------------------------------------------------------
// Conversions
// ---------------------------------------------------------------------------
__global__ void egat_cast_bf16(const float* __restrict__ src, bf16* __restrict__ dst, int n) {
  for (int i = blockIdx.x * blockDim.x + threadIdx.x; i < n; i += gridDim.x * blockDim.x)
    dst[i] = (bf16)src[i];
}

// W [H,D,E] f32 -> WT [H,E,D] bf16
__global__ void egat_transpose_w(const float* __restrict__ W, bf16* __restrict__ WT) {
  const int total = Hn * Dn * En;
  for (int i = blockIdx.x * blockDim.x + threadIdx.x; i < total; i += gridDim.x * blockDim.x) {
    int e = i % En;
    int d = (i / En) % Dn;
    int h = i / (Dn * En);
    WT[((size_t)h * En + e) * Dn + d] = (bf16)W[i];
  }
}

// ---------------------------------------------------------------------------
// Projection GEMM: h[b,h] (512x96) = feat[b] (512x768) x W[h] (768x96)
// One wave per 16-row stripe; 6 accumulators cover E=96.
// mode 0: store f32 [B,H,N,E]; mode 1: store bf16 transposed [B,H,E,N]
// ---------------------------------------------------------------------------
__global__ void egat_proj_gemm(const bf16* __restrict__ featb, const bf16* __restrict__ wT,
                               float* __restrict__ out_f32, bf16* __restrict__ outT_bf, int mode) {
  const int lane = threadIdx.x & 31;
  const int gw   = blockIdx.x * (blockDim.x >> 5) + (threadIdx.x >> 5); // B*H*32 waves
  const int mt = gw & 31;
  const int h  = (gw >> 5) & (Hn - 1);
  const int b  = gw >> 8;
  const int m0 = mt * 16;

  const bf16* A  = featb + (size_t)b * Nn * Dn;
  const bf16* BT = wT + (size_t)h * En * Dn;

  v8f acc[6] = {};
  for (int k0 = 0; k0 < Dn; k0 += 32) {
    v16bf a = load_a_frag(A, Dn, m0, k0, lane);
    __builtin_prefetch(A + (size_t)(m0 + (lane & 15)) * Dn + k0 + 64, 0, 1);
#pragma unroll
    for (int j = 0; j < 6; ++j) {
      v16bf bb = load_b_frag(BT, Dn, j * 16, k0, lane);
      acc[j] = wmma_bf16(a, bb, acc[j]);
    }
  }

  const int Mr = (lane >> 4) * 8;
  const int Nc = lane & 15;
#pragma unroll
  for (int j = 0; j < 6; ++j) {
#pragma unroll
    for (int r = 0; r < 8; ++r) {
      const int m = m0 + Mr + r;
      const int e = j * 16 + Nc;
      const float v = acc[j][r];
      if (mode == 0)
        out_f32[((size_t)(b * Hn + h) * Nn + m) * En + e] = v;
      else
        outT_bf[((size_t)(b * Hn + h) * En + e) * Nn + m] = (bf16)v;
    }
  }
}

// ---------------------------------------------------------------------------
// a_src / a_dst: t = tanh(h1[b,h,n,:]); dot with w_src[h], w_dst[h]
// ---------------------------------------------------------------------------
__global__ void egat_attn_vec(const float* __restrict__ h1, const float* __restrict__ wsrc,
                              const float* __restrict__ wdst, float* __restrict__ asrc,
                              float* __restrict__ adst) {
  const int i = blockIdx.x * blockDim.x + threadIdx.x; // over B*H*N
  if (i >= Bn * Hn * Nn) return;
  const int h = (i / Nn) % Hn;
  const float* row = h1 + (size_t)i * En;
  float s1 = 0.f, s2 = 0.f;
  for (int e = 0; e < En; ++e) {
    const float t = tanhf(row[e]);
    s1 += t * wsrc[h * En + e];
    s2 += t * wdst[h * En + e];
  }
  asrc[i] = s1;
  adst[i] = s2;
}

// ---------------------------------------------------------------------------
// Per-row masked double softmax + attn-map output. One 256-thread block/row.
// ---------------------------------------------------------------------------
__device__ __forceinline__ float blk_red_max(float v, float* sm) {
  const int t = threadIdx.x;
  sm[t] = v; __syncthreads();
#pragma unroll
  for (int s = 128; s > 0; s >>= 1) {
    if (t < s) sm[t] = fmaxf(sm[t], sm[t + s]);
    __syncthreads();
  }
  const float r = sm[0]; __syncthreads();
  return r;
}
__device__ __forceinline__ float blk_red_sum(float v, float* sm) {
  const int t = threadIdx.x;
  sm[t] = v; __syncthreads();
#pragma unroll
  for (int s = 128; s > 0; s >>= 1) {
    if (t < s) sm[t] += sm[t + s];
    __syncthreads();
  }
  const float r = sm[0]; __syncthreads();
  return r;
}

__global__ void egat_softmax(const int* __restrict__ adj, const int* __restrict__ smask,
                             const float* __restrict__ asrc, const float* __restrict__ adst,
                             bf16* __restrict__ attn2b, bf16* __restrict__ attn3b,
                             float* __restrict__ attn_out) {
  __shared__ float sm[256];
  const int blk = blockIdx.x;            // over B*H*N rows
  const int i = blk % Nn;
  const int h = (blk / Nn) % Hn;
  const int b = blk / (Nn * Hn);
  const int t = threadIdx.x;

  const int* arow = adj   + ((size_t)b * Nn + i) * Nn;
  const int* srow = smask + ((size_t)b * Nn + i) * Nn;
  const float as = asrc[((size_t)b * Hn + h) * Nn + i];
  const float* adrow = adst + ((size_t)b * Hn + h) * Nn;

  float x[2]; int m2[2], m3[2];
#pragma unroll
  for (int q = 0; q < 2; ++q) {
    const int j = t + q * 256;
    const int a = arow[j], s = srow[j];
    m2[q] = a & s;          // inter: adj where same-speaker
    m3[q] = a & (1 - s);    // outer: adj where diff-speaker
    const float v = as + adrow[j];
    x[q] = v > 0.f ? v : LEAKY * v;
  }
  const float max2 = blk_red_max(fmaxf(m2[0] ? x[0] : NEGV, m2[1] ? x[1] : NEGV), sm);
  const float max3 = blk_red_max(fmaxf(m3[0] ? x[0] : NEGV, m3[1] ? x[1] : NEGV), sm);

  float e2[2], e3[2], l2 = 0.f, l3 = 0.f;
#pragma unroll
  for (int q = 0; q < 2; ++q) {
    e2[q] = m2[q] ? __expf(x[q] - max2) : 0.f; l2 += e2[q];
    e3[q] = m3[q] ? __expf(x[q] - max3) : 0.f; l3 += e3[q];
  }
  const float sum2 = blk_red_sum(l2, sm);
  const float sum3 = blk_red_sum(l3, sm);
  const float r2 = sum2 > 0.f ? 1.f / sum2 : 0.f;
  const float r3 = sum3 > 0.f ? 1.f / sum3 : 0.f;

  const size_t base = (((size_t)b * Hn + h) * Nn + i) * Nn;
#pragma unroll
  for (int q = 0; q < 2; ++q) {
    const int j = t + q * 256;
    const float p2 = e2[q] * r2;
    const float p3 = e3[q] * r3;
    attn2b[base + j] = (bf16)p2;
    attn3b[base + j] = (bf16)p3;
    attn_out[base + j] = (j == i ? 1.f : 0.f) + p2 + p3;
  }
}

// ---------------------------------------------------------------------------
// Aggregation GEMM: featpre = attn2 @ h2 + attn3 @ h3 + diag(adj)*h1 + bias
// ---------------------------------------------------------------------------
__global__ void egat_agg_gemm(const bf16* __restrict__ attn2b, const bf16* __restrict__ attn3b,
                              const bf16* __restrict__ h2T, const bf16* __restrict__ h3T,
                              const float* __restrict__ h1, const int* __restrict__ adj,
                              const float* __restrict__ bias, float* __restrict__ featpre) {
  const int lane = threadIdx.x & 31;
  const int gw   = blockIdx.x * (blockDim.x >> 5) + (threadIdx.x >> 5); // B*H*32
  const int mt = gw & 31;
  const int h  = (gw >> 5) & (Hn - 1);
  const int b  = gw >> 8;
  const int m0 = mt * 16;

  const size_t bh = (size_t)b * Hn + h;
  const bf16* A2 = attn2b + bh * Nn * Nn;
  const bf16* A3 = attn3b + bh * Nn * Nn;
  const bf16* B2 = h2T + bh * En * Nn;
  const bf16* B3 = h3T + bh * En * Nn;

  v8f acc[6] = {};
  for (int k0 = 0; k0 < Nn; k0 += 32) {
    v16bf a = load_a_frag(A2, Nn, m0, k0, lane);
#pragma unroll
    for (int j = 0; j < 6; ++j) acc[j] = wmma_bf16(a, load_b_frag(B2, Nn, j * 16, k0, lane), acc[j]);
  }
  for (int k0 = 0; k0 < Nn; k0 += 32) {
    v16bf a = load_a_frag(A3, Nn, m0, k0, lane);
#pragma unroll
    for (int j = 0; j < 6; ++j) acc[j] = wmma_bf16(a, load_b_frag(B3, Nn, j * 16, k0, lane), acc[j]);
  }

  const int Mr = (lane >> 4) * 8;
  const int Nc = lane & 15;
#pragma unroll
  for (int j = 0; j < 6; ++j) {
#pragma unroll
    for (int r = 0; r < 8; ++r) {
      const int m = m0 + Mr + r;
      const int e = j * 16 + Nc;
      const float diag = (float)adj[((size_t)b * Nn + m) * Nn + m]; // intra @ h1 == row scale
      featpre[(bh * Nn + m) * En + e] =
          acc[j][r] + diag * h1[(bh * Nn + m) * En + e] + bias[e];
    }
  }
}

// ---------------------------------------------------------------------------
// Gate GEMM: gatelin[B*N, D] = feat[B*N, D] x Hw^T  (BT row-major == Hw)
// ---------------------------------------------------------------------------
__global__ void egat_gate_gemm(const bf16* __restrict__ featb, const bf16* __restrict__ hwT,
                               float* __restrict__ gatelin) {
  const int lane = threadIdx.x & 31;
  const int gw   = blockIdx.x * (blockDim.x >> 5) + (threadIdx.x >> 5); // 2048 waves
  const int ng = gw & 7;        // 8 groups of 96 cols
  const int mt = gw >> 3;       // 256 row tiles
  const int m0 = mt * 16;
  const int n_base = ng * 96;

  v8f acc[6] = {};
  for (int k0 = 0; k0 < Dn; k0 += 32) {
    v16bf a = load_a_frag(featb, Dn, m0, k0, lane);
#pragma unroll
    for (int j = 0; j < 6; ++j)
      acc[j] = wmma_bf16(a, load_b_frag(hwT, Dn, n_base + j * 16, k0, lane), acc[j]);
  }

  const int Mr = (lane >> 4) * 8;
  const int Nc = lane & 15;
#pragma unroll
  for (int j = 0; j < 6; ++j)
#pragma unroll
    for (int r = 0; r < 8; ++r)
      gatelin[(size_t)(m0 + Mr + r) * Dn + n_base + j * 16 + Nc] = acc[j][r];
}

// ---------------------------------------------------------------------------
// Final fuse: out = sigmoid(gatelin+Hb) * elu(featpre) + (1-g) * feat_in
// ---------------------------------------------------------------------------
__global__ void egat_final(const float* __restrict__ featpre, const float* __restrict__ gatelin,
                           const float* __restrict__ hb, const float* __restrict__ feat_in,
                           float* __restrict__ out) {
  const int total = Bn * Nn * Dn;
  for (int idx = blockIdx.x * blockDim.x + threadIdx.x; idx < total; idx += gridDim.x * blockDim.x) {
    const int d  = idx % Dn;
    const int n  = (idx / Dn) % Nn;
    const int b  = idx / (Dn * Nn);
    const int h  = d / En;
    const int e  = d % En;
    const float fp = featpre[(((size_t)b * Hn + h) * Nn + n) * En + e];
    const float fe = fp > 0.f ? fp : (__expf(fp) - 1.f); // elu
    const float g  = 1.f / (1.f + __expf(-(gatelin[idx] + hb[d])));
    out[idx] = g * fe + (1.f - g) * feat_in[idx];
  }
}

// ---------------------------------------------------------------------------
extern "C" void kernel_launch(void* const* d_in, const int* in_sizes, int n_in,
                              void* d_out, int out_size, void* d_ws, size_t ws_size,
                              hipStream_t stream) {
  (void)in_sizes; (void)n_in; (void)out_size; (void)ws_size;

  const float* feat_in = (const float*)d_in[0];
  const int*   adj     = (const int*)d_in[1];
  const int*   smask   = (const int*)d_in[2];
  const float* W1      = (const float*)d_in[3];
  const float* W2      = (const float*)d_in[4];
  const float* W3      = (const float*)d_in[5];
  const float* bias    = (const float*)d_in[6];
  const float* w_src   = (const float*)d_in[7];
  const float* w_dst   = (const float*)d_in[8];
  const float* Hw      = (const float*)d_in[9];
  const float* Hb      = (const float*)d_in[10];

  char* ws = (char*)d_ws;
  size_t off = 0;
  auto carve = [&](size_t bytes) -> char* {
    char* p = ws + off;
    off = (off + bytes + 255) & ~(size_t)255;
    return p;
  };

  const size_t NE  = (size_t)Bn * Hn * Nn * En;   // 3,145,728
  const size_t NN  = (size_t)Bn * Hn * Nn * Nn;   // 16,777,216
  const size_t BND = (size_t)Bn * Nn * Dn;        // 3,145,728
  const size_t WSZ = (size_t)Hn * En * Dn;        // 589,824

  bf16*  featb  = (bf16*)carve(BND * 2);
  bf16*  w1T    = (bf16*)carve(WSZ * 2);
  bf16*  w2T    = (bf16*)carve(WSZ * 2);
  bf16*  w3T    = (bf16*)carve(WSZ * 2);
  bf16*  hwT    = (bf16*)carve((size_t)Dn * Dn * 2);
  float* h1     = (float*)carve(NE * 4);
  bf16*  h2T    = (bf16*)carve(NE * 2);
  bf16*  h3T    = (bf16*)carve(NE * 2);
  float* asrc   = (float*)carve((size_t)Bn * Hn * Nn * 4);
  float* adst   = (float*)carve((size_t)Bn * Hn * Nn * 4);
  bf16*  attn2b = (bf16*)carve(NN * 2);
  bf16*  attn3b = (bf16*)carve(NN * 2);
  float* featpre= (float*)carve(NE * 4);
  float* gatelin= (float*)carve(BND * 4);

  float* out1 = (float*)d_out;          // [B,N,D]
  float* out2 = out1 + BND;             // [B,H,N,N]

  // 1) conversions
  egat_cast_bf16<<<2048, 256, 0, stream>>>(feat_in, featb, (int)BND);
  egat_cast_bf16<<<1024, 256, 0, stream>>>(Hw, hwT, Dn * Dn);
  egat_transpose_w<<<2048, 256, 0, stream>>>(W1, w1T);
  egat_transpose_w<<<2048, 256, 0, stream>>>(W2, w2T);
  egat_transpose_w<<<2048, 256, 0, stream>>>(W3, w3T);

  // 2) projections: 2048 waves, 4 waves/block
  egat_proj_gemm<<<512, 128, 0, stream>>>(featb, w1T, h1, nullptr, 0);
  egat_proj_gemm<<<512, 128, 0, stream>>>(featb, w2T, nullptr, h2T, 1);
  egat_proj_gemm<<<512, 128, 0, stream>>>(featb, w3T, nullptr, h3T, 1);

  // 3) attention logits
  egat_attn_vec<<<(Bn * Hn * Nn + 255) / 256, 256, 0, stream>>>(h1, w_src, w_dst, asrc, adst);

  // 4) dual masked softmax + attention-map output
  egat_softmax<<<Bn * Hn * Nn, 256, 0, stream>>>(adj, smask, asrc, adst, attn2b, attn3b, out2);

  // 5) aggregation GEMM (+ diagonal intra term + bias)
  egat_agg_gemm<<<512, 128, 0, stream>>>(attn2b, attn3b, h2T, h3T, h1, adj, bias, featpre);

  // 6) gate GEMM
  egat_gate_gemm<<<512, 128, 0, stream>>>(featb, hwT, gatelin);

  // 7) fuse elu / sigmoid / residual mix
  egat_final<<<2048, 256, 0, stream>>>(featpre, gatelin, Hb, feat_in, out1);
}
